// PlaneNet_82240033783809
// MI455X (gfx1250) — matrix operand order repaired and checked by reading, not compiled, not run on hardware
//
#include <hip/hip_runtime.h>
#include <hip/hip_fp16.h>

typedef __attribute__((ext_vector_type(16))) _Float16 v16h;
typedef __attribute__((ext_vector_type(8)))  float    v8f;
typedef __attribute__((ext_vector_type(4)))  float    f32x4;

#define PP 3
#define NN 50000
#define EE 256000
#define CC 5
#define FF 68          // per-class node feature dim
#define F2 136         // 2*FF (concat)
#define FE 16          // edge hidden
#define FN 64          // node hidden / output

// Pre-packed weight fragment table sizes (halves). Each fragment = 32 lanes x 16 halves.
#define FRAG_E1 (PP * CC * 5 * 512)            // We1: 1 row-tile x 5 ksteps
#define FRAG_N1 (PP * CC * 4 * 5 * 512)        // Wn1: 4 row-tiles x 5 ksteps
#define FRAG_N2 (PP * CC * 4 * 2 * 512)        // Wn2: 4 row-tiles x 2 ksteps
#define FRAG_TOT (FRAG_E1 + FRAG_N1 + FRAG_N2) // 253,440 halves

// ---- WMMA 16x16x32 f16 fragment index maps (contiguous k order, same for A and B) ----
// A (16xK tile, M = lane&15): lanes<16 hold K {0..15}, lanes>=16 hold K {16..31}
__device__ __forceinline__ int a_kloc(int lane, int j) {
    return j + ((lane & 16) ? 16 : 0);
}

__device__ __forceinline__ void atomAddF(float* p, float v) {
    unsafeAtomicAdd(p, v);   // -> global_atomic_add_f32 (no return)
}

__device__ __forceinline__ v16h cvt16(f32x4 a, f32x4 b, f32x4 c, f32x4 d) {
    v16h r;
#pragma unroll
    for (int i = 0; i < 4; ++i) {
        r[i]      = (_Float16)a[i];
        r[4 + i]  = (_Float16)b[i];
        r[8 + i]  = (_Float16)c[i];
        r[12 + i] = (_Float16)d[i];
    }
    return r;
}

// =====================================================================
// Branchless B-fragment builder for a concatenated row [rowA(68) ; rowB(68)]
// padded with zeros to K=160.  B layout: N = lane&15; lanes<16 hold
// K kb..kb+15, lanes>=16 hold K kb+16..kb+31.  All loads are aligned
// f32x4 vector loads issued unconditionally; boundary chunks resolved
// with v_cndmask value selects (no exec-mask divergence).
// =====================================================================
__device__ __forceinline__ void build_b_frags(const float* __restrict__ rowA,
                                              const float* __restrict__ rowB,
                                              bool hi, v16h bf[5])
{
    // s=0: K 0..31   (pure rowA; lane offset 0 / 16)
    {
        const f32x4* p = (const f32x4*)(rowA + (hi ? 16 : 0));
        bf[0] = cvt16(p[0], p[1], p[2], p[3]);
    }
    // s=1: K 32..63  (pure rowA; offset 32 / 48)
    {
        const f32x4* p = (const f32x4*)(rowA + (hi ? 48 : 32));
        bf[1] = cvt16(p[0], p[1], p[2], p[3]);
    }
    // s=2: K 64..95  (lo: rowA[64..67]+rowB[0..11]; hi: rowB[12..27])
    {
        f32x4 xq = *(const f32x4*)(rowA + 64);
        const float* bp = rowB + (hi ? 12 : 0);
        f32x4 a0 = *(const f32x4*)(bp);
        f32x4 a1 = *(const f32x4*)(bp + 4);
        f32x4 a2 = *(const f32x4*)(bp + 8);
        f32x4 a3 = *(const f32x4*)(bp + 12);
        v16h r;
#pragma unroll
        for (int i = 0; i < 4; ++i) {
            r[i]      = (_Float16)(hi ? a0[i] : xq[i]);
            r[4 + i]  = (_Float16)(hi ? a1[i] : a0[i]);
            r[8 + i]  = (_Float16)(hi ? a2[i] : a1[i]);
            r[12 + i] = (_Float16)(hi ? a3[i] : a2[i]);
        }
        bf[2] = r;
    }
    // s=3: K 96..127 (pure rowB; offset 28 / 44)
    {
        const f32x4* p = (const f32x4*)(rowB + (hi ? 44 : 28));
        bf[3] = cvt16(p[0], p[1], p[2], p[3]);
    }
    // s=4: K 128..159 (lo: rowB[60..67] then zero pad; hi: all zero)
    {
        f32x4 z0 = *(const f32x4*)(rowB + 60);
        f32x4 z1 = *(const f32x4*)(rowB + 64);
        v16h r;
#pragma unroll
        for (int i = 0; i < 4; ++i) {
            r[i]      = (_Float16)(hi ? 0.f : z0[i]);
            r[4 + i]  = (_Float16)(hi ? 0.f : z1[i]);
            r[8 + i]  = (_Float16)0.f;
            r[12 + i] = (_Float16)0.f;
        }
        bf[4] = r;
    }
}

// =====================================================================
// Prep: pack We1/Wn1/Wn2 into f16 WMMA A-fragment layout (zero-padded
// K). One thread per output half. Layout order: [E1 | N1 | N2], each
// fragment = 512 halves, lane-major (lane*16 + j).
// =====================================================================
__global__ __launch_bounds__(256)
void prep_weights(const float* __restrict__ We1, const float* __restrict__ Wn1,
                  const float* __restrict__ Wn2, _Float16* __restrict__ wf)
{
    int idx = blockIdx.x * 256 + threadIdx.x;
    if (idx >= FRAG_TOT) return;
    float val = 0.f;
    if (idx < FRAG_E1) {
        int j = idx & 15, lane = (idx >> 4) & 31;
        int fi = idx >> 9;                 // pc*5 + s
        int s = fi % 5, pc = fi / 5;
        int row = lane & 15;
        int ka = s * 32 + a_kloc(lane, j);
        if (ka < F2) val = We1[(size_t)pc * FE * F2 + row * F2 + ka];
    } else if (idx < FRAG_E1 + FRAG_N1) {
        int i2 = idx - FRAG_E1;
        int j = i2 & 15, lane = (i2 >> 4) & 31;
        int fi = i2 >> 9;                  // (pc*4 + t)*5 + s
        int s = fi % 5, t = (fi / 5) & 3, pc = fi / 20;
        int row = t * 16 + (lane & 15);
        int ka = s * 32 + a_kloc(lane, j);
        if (ka < F2) val = Wn1[(size_t)pc * FN * F2 + row * F2 + ka];
    } else {
        int i3 = idx - FRAG_E1 - FRAG_N1;
        int j = i3 & 15, lane = (i3 >> 4) & 31;
        int fi = i3 >> 9;                  // (pc*4 + t)*2 + s
        int s = fi & 1, t = (fi >> 1) & 3, pc = fi >> 3;
        int row = t * 16 + (lane & 15);
        int ka = s * 32 + a_kloc(lane, j); // always < 64
        val = Wn2[(size_t)pc * FN * FN + row * FN + ka];
    }
    wf[idx] = (_Float16)val;
}

// =====================================================================
// Edge kernel: per 16-edge tile, per class: h = tanh(We1 @ [x_i;x_j]),
// logit = We2 . h + be2; softmax over classes; scatter w*x_j into aggr.
// =====================================================================
__global__ __launch_bounds__(256)
void edge_kernel(const float* __restrict__ x, const int* __restrict__ ei,
                 const _Float16* __restrict__ wf,
                 const float* __restrict__ be1,
                 const float* __restrict__ We2, const float* __restrict__ be2,
                 float* __restrict__ aggr)
{
    const int lane = threadIdx.x & 31;
    const int wave = threadIdx.x >> 5;
    const int tilesPerPlane = EE / 16;                 // 16000
    const int tile = blockIdx.x * 8 + wave;
    if (tile >= PP * tilesPerPlane) return;            // wave-uniform
    const int p  = tile / tilesPerPlane;
    const int e0 = (tile % tilesPerPlane) * 16;
    const int n  = lane & 15;                          // edge column in tile
    const int e  = e0 + n;

    const int src = ei[(size_t)p * 2 * EE + e];
    const int dst = ei[(size_t)p * 2 * EE + EE + e];
    const float* xp = x + (size_t)p * NN * CC * FF;
    const float* xi = xp + (size_t)dst * CC * FF;      // target (x_i)
    const float* xj = xp + (size_t)src * CC * FF;      // source (x_j)

    // warm L2/L0 for the first class's gathers
    __builtin_prefetch(xi, 0, 0);
    __builtin_prefetch(xj, 0, 0);

    const int mofs = (lane & 16) ? 8 : 0;
    const bool hi = (lane & 16) != 0;
    float logit[CC];

    for (int c = 0; c < CC; ++c) {
        if (c + 1 < CC) {                              // overlap next gather with WMMA
            __builtin_prefetch(xi + (c + 1) * FF, 0, 0);
            __builtin_prefetch(xj + (c + 1) * FF, 0, 0);
        }
        v16h bf[5];
        build_b_frags(xi + c * FF, xj + c * FF, hi, bf);

        const _Float16* wfa = wf + (size_t)(p * CC + c) * 5 * 512 + lane * 16;
        v8f acc = {0.f, 0.f, 0.f, 0.f, 0.f, 0.f, 0.f, 0.f};
#pragma unroll
        for (int s = 0; s < 5; ++s) {                  // K = 136 padded to 160
            v16h a = *(const v16h*)(wfa + s * 512);
            acc = __builtin_amdgcn_wmma_f32_16x16x32_f16(
                false, a, false, bf[s], (short)0, acc, false, false);
        }
        // D layout: acc[r] = h_pre[m][n], m = r + mofs
        const float* b1 = be1 + (size_t)(p * CC + c) * FE;
        const float* w2 = We2 + (size_t)(p * CC + c) * FE;
        float part = 0.f;
#pragma unroll
        for (int r = 0; r < 8; ++r) {
            int m = r + mofs;
            part += tanhf(acc[r] + b1[m]) * w2[m];
        }
        part += __shfl_xor(part, 16, 32);              // combine row halves
        logit[c] = part + be2[p * CC + c];
    }

    // softmax over the 5 classes (per edge; duplicated in lanes n and n+16)
    float mx = logit[0];
#pragma unroll
    for (int c = 1; c < CC; ++c) mx = fmaxf(mx, logit[c]);
    float w[CC], sum = 0.f;
#pragma unroll
    for (int c = 0; c < CC; ++c) { w[c] = __expf(logit[c] - mx); sum += w[c]; }
    const float inv = 1.f / sum;

    // scatter-add msg = w * x_j into aggr[dst]; lane covers 34 of 68 features
    const int f0 = hi ? 34 : 0;
    float* ag = aggr + ((size_t)p * NN + dst) * CC * FF;
    for (int c = 0; c < CC; ++c) {
        const float wc = w[c] * inv;
        const float* xjc = xj + c * FF;
        float* agc = ag + c * FF;
#pragma unroll
        for (int f = f0; f < f0 + 34; ++f)
            atomAddF(&agc[f], wc * xjc[f]);
    }
}

// =====================================================================
// Node kernel: per 16-node tile, per class:
//   h2  = tanh(Wn1 @ [x; aggr] + bn1)   (64x136 -> 4x5 WMMAs)
//   out = tanh(Wn2 @ h2 + bn2)          (64x64  -> 4x2 WMMAs)
// D->B relayout of h2 done in-register via shfl_xor(16).
// =====================================================================
__global__ __launch_bounds__(256)
void node_kernel(const float* __restrict__ x, const float* __restrict__ aggr,
                 const _Float16* __restrict__ wf,
                 const float* __restrict__ bn1,
                 const float* __restrict__ bn2,
                 float* __restrict__ out)
{
    const int lane = threadIdx.x & 31;
    const int wave = threadIdx.x >> 5;
    const int tilesPerPlane = NN / 16;                 // 3125
    const int tile = blockIdx.x * 8 + wave;
    if (tile >= PP * tilesPerPlane) return;            // wave-uniform
    const int p  = tile / tilesPerPlane;
    const int n0 = (tile % tilesPerPlane) * 16;
    const int n  = lane & 15;
    const int node = n0 + n;

    const float* xc = x    + ((size_t)p * NN + node) * CC * FF;
    const float* ac = aggr + ((size_t)p * NN + node) * CC * FF;
    const int mofs = (lane & 16) ? 8 : 0;
    const bool hi = (lane & 16) != 0;
    const _Float16* wfN1 = wf + FRAG_E1;
    const _Float16* wfN2 = wf + FRAG_E1 + FRAG_N1;

    for (int c = 0; c < CC; ++c) {
        const float* b1 = bn1 + (size_t)(p * CC + c) * FN;
        const float* b2 = bn2 + (size_t)(p * CC + c) * FN;

        // B fragments of u_in = [x ; aggr] (shared by all 4 row tiles)
        v16h bf[5];
        build_b_frags(xc + c * FF, ac + c * FF, hi, bf);

        // Layer 1: h2 = tanh(Wn1 @ u_in + bn1), rows 0..63 in 4 tiles
        float h2v[4][8];
#pragma unroll
        for (int t = 0; t < 4; ++t) {
            const _Float16* wfa =
                wfN1 + (size_t)((p * CC + c) * 4 + t) * 5 * 512 + lane * 16;
            v8f acc = {0.f, 0.f, 0.f, 0.f, 0.f, 0.f, 0.f, 0.f};
#pragma unroll
            for (int s = 0; s < 5; ++s) {
                v16h a = *(const v16h*)(wfa + s * 512);
                acc = __builtin_amdgcn_wmma_f32_16x16x32_f16(
                    false, a, false, bf[s], (short)0, acc, false, false);
            }
#pragma unroll
            for (int r = 0; r < 8; ++r)
                h2v[t][r] = tanhf(acc[r] + b1[t * 16 + mofs + r]);
        }

        // Relayout h2 (D layout) -> B fragments (K = 64, 2 ksteps), in-register
        v16h hb[2];
#pragma unroll
        for (int s = 0; s < 2; ++s) {
#pragma unroll
            for (int r = 0; r < 8; ++r) {
                float o0 = __shfl_xor(h2v[2 * s][r], 16, 32);
                float o1 = __shfl_xor(h2v[2 * s + 1][r], 16, 32);
                hb[s][r]     = (_Float16)(hi ? o1 : h2v[2 * s][r]);     // j = r
                hb[s][8 + r] = (_Float16)(hi ? h2v[2 * s + 1][r] : o0); // j = 8+r
            }
        }

        // Layer 2: out = tanh(Wn2 @ h2 + bn2)
        float* oc = out + (((size_t)p * NN + node) * CC + c) * FN;
#pragma unroll
        for (int t = 0; t < 4; ++t) {
            const _Float16* wfa =
                wfN2 + (size_t)((p * CC + c) * 4 + t) * 2 * 512 + lane * 16;
            v8f acc = {0.f, 0.f, 0.f, 0.f, 0.f, 0.f, 0.f, 0.f};
#pragma unroll
            for (int s = 0; s < 2; ++s) {
                v16h a = *(const v16h*)(wfa + s * 512);
                acc = __builtin_amdgcn_wmma_f32_16x16x32_f16(
                    false, a, false, hb[s], (short)0, acc, false, false);
            }
#pragma unroll
            for (int r = 0; r < 8; ++r) {
                int m = t * 16 + mofs + r;
                oc[m] = tanhf(acc[r] + b2[m]);   // 8 contiguous floats per lane
            }
        }
    }
}

extern "C" void kernel_launch(void* const* d_in, const int* in_sizes, int n_in,
                              void* d_out, int out_size, void* d_ws, size_t ws_size,
                              hipStream_t stream)
{
    (void)in_sizes; (void)n_in; (void)out_size;
    const float* x   = (const float*)d_in[0];
    const int*   ei  = (const int*)  d_in[1];   // int64 in ref demotes to int32 under jax defaults
    const float* We1 = (const float*)d_in[2];
    const float* be1 = (const float*)d_in[3];
    const float* We2 = (const float*)d_in[4];
    const float* be2 = (const float*)d_in[5];
    const float* Wn1 = (const float*)d_in[6];
    const float* bn1 = (const float*)d_in[7];
    const float* Wn2 = (const float*)d_in[8];
    const float* bn2 = (const float*)d_in[9];
    float* out  = (float*)d_out;

    float* aggr = (float*)d_ws;                 // [P, N, C, F] f32 scratch
    const size_t aggrBytes = (size_t)PP * NN * CC * FF * sizeof(float);  // 204 MB, 32B-mult
    _Float16* wf = (_Float16*)((char*)d_ws + aggrBytes);
    const size_t fragBytes = (size_t)FRAG_TOT * sizeof(_Float16);
    if (ws_size < aggrBytes + fragBytes) return; // defensive

    hipMemsetAsync(aggr, 0, aggrBytes, stream); // deterministic re-zero each launch

    prep_weights<<<(FRAG_TOT + 255) / 256, 256, 0, stream>>>(We1, Wn1, Wn2, wf);

    const int edgeTiles = PP * (EE / 16);       // 48000 -> 6000 blocks of 8 waves
    edge_kernel<<<edgeTiles / 8, 256, 0, stream>>>(x, ei, wf, be1, We2, be2, aggr);

    const int nodeTiles = PP * (NN / 16);       // 9375
    node_kernel<<<(nodeTiles + 7) / 8, 256, 0, stream>>>(x, aggr, wf, bn1, bn2, out);
}
